// SelfAttention_23579370455641
// MI455X (gfx1250) — matrix operand order repaired and checked by reading, not compiled, not run on hardware
//
#include <hip/hip_runtime.h>
#include <hip/hip_bf16.h>

typedef __attribute__((ext_vector_type(16))) __bf16 v16bf;
typedef __attribute__((ext_vector_type(8)))  float  v8f;

#define N_POS 4096
#define C_DIM 256
#define A_DIM 32
#define SS    4100                      // padded f32 score row stride (floats)
#define SSB   4104                      // padded bf16 attn row stride (bf16)
#define OUT_ELEMS (8ull * 256 * 4096)   // main output size, attention follows

// ---------------- projection: 32 output channels per block ----------------
// grid.x = 128 (b, n-tile of 256), grid.y = #chunks of 32 rows, block = 256
__global__ __launch_bounds__(256) void proj32_kernel(
    const float* __restrict__ x, const float* __restrict__ W,
    const float* __restrict__ bias, __bf16* __restrict__ dst, int transposed)
{
  __shared__ float sW[32 * 256];        // 32 KB: W chunk, stored [c][a]
  const int tid = threadIdx.x;
  const int b   = blockIdx.x >> 4;
  const int n   = (blockIdx.x & 15) * 256 + tid;
  const int rowBase = blockIdx.y * 32;

  for (int i = tid; i < 32 * 256; i += 256) {
    int c = i & 255, a = i >> 8;        // consecutive i -> consecutive c: coalesced
    sW[c * 32 + a] = W[(size_t)(rowBase + a) * 256 + c];
  }
  __syncthreads();

  float acc[32];
#pragma unroll
  for (int a = 0; a < 32; ++a) acc[a] = 0.f;

  const float* xb = x + (size_t)b * C_DIM * N_POS + n;
  for (int c = 0; c < 256; ++c) {
    float xv = xb[(size_t)c * N_POS];   // coalesced across threads (contiguous n)
    const float* w = sW + c * 32;       // LDS broadcast
#pragma unroll
    for (int a = 0; a < 32; ++a) acc[a] += xv * w[a];
  }

  if (transposed) {                     // V: store [b][c][n] (n contiguous)
#pragma unroll
    for (int a = 0; a < 32; ++a)
      dst[((size_t)b * C_DIM + rowBase + a) * N_POS + n] =
          (__bf16)(acc[a] + bias[rowBase + a]);
  } else {                              // Q/K: store [b][n][a] (a contiguous)
#pragma unroll
    for (int a = 0; a < 32; ++a)
      dst[((size_t)b * N_POS + n) * A_DIM + a] = (__bf16)(acc[a] + bias[a]);
  }
}

// group-of-32 XOR swizzle max over the 8 running row maxima
#define SWIZ_MAX(pat)                                                        \
  {                                                                          \
    _Pragma("unroll")                                                        \
    for (int r = 0; r < 8; ++r) {                                            \
      int o = __builtin_amdgcn_ds_swizzle(__float_as_int(rmax[r]), (pat));   \
      rmax[r] = fmaxf(rmax[r], __int_as_float(o));                           \
    }                                                                        \
  }

// ---------------- fused attention: one block per (batch, 16-row q-tile) ----
// grid.x = 8*256 = 2048, block = 256 (8 waves), dynamic LDS ~259 KB
__global__ __launch_bounds__(256) void attn_kernel(
    const float* __restrict__ x, const __bf16* __restrict__ Qb,
    const __bf16* __restrict__ Kb, const __bf16* __restrict__ Vt,
    const float* __restrict__ gammaPtr,
    float* __restrict__ out, float* __restrict__ attnOut)
{
  extern __shared__ char smem[];
  float*  sS   = (float*)smem;                          // 16 x SS scores (f32)
  __bf16* sB   = (__bf16*)smem;                         // 16 x SSB attn (bf16), aliases sS bottom
  __bf16* sQ   = (__bf16*)(smem + 16 * SS * 4);         // 16 x 32 Q tile
  float*  sRed = (float*)(smem + 16 * SS * 4 + 1024);   // 16 rows x 16 partials
  float*  sMax = sRed + 256;                            // 16 row maxima
  float*  sInv = sMax + 16;                             // 16 inverse sums

  const int tid  = threadIdx.x;
  const int lane = tid & 31;
  const int wave = tid >> 5;
  const int half = lane >> 4;
  const int l16  = lane & 15;
  const int b    = blockIdx.x >> 8;
  const int qt   = blockIdx.x & 255;
  const int q0   = qt * 16;

  // Phase 1: Q tile -> LDS
  for (int i = tid; i < 16 * 32; i += 256) {
    int r = i >> 5, a = i & 31;
    sQ[i] = Qb[((size_t)b * N_POS + q0 + r) * A_DIM + a];
  }
  __syncthreads();

  // A-fragment of Q (16x32 bf16), contiguous k = 16*half + e (2x ds_load_b128)
  v16bf aQ;
  {
    const __bf16* qr = sQ + l16 * 32;
#pragma unroll
    for (int e = 0; e < 16; ++e) {
      int k = half * 16 + e;
      aQ[e] = qr[k];
    }
  }

  // Phase 2: scores(16 x 4096) = Q @ K^T; one WMMA per 16-key tile (K-dim=32).
  // 2-way software pipeline: two independent WMMAs back-to-back, then the
  // fold/store VALU of one fills the WMMA->VALU hazard window of the other.
  float rmax[8];
#pragma unroll
  for (int r = 0; r < 8; ++r) rmax[r] = -3.4e38f;

  for (int kt = wave; kt < 256; kt += 16) {
    const int kt1 = kt + 8;
    const v16bf* bp0 =
        (const v16bf*)(Kb + ((size_t)b * N_POS + kt  * 16 + l16) * A_DIM + half * 16);
    const v16bf* bp1 =
        (const v16bf*)(Kb + ((size_t)b * N_POS + kt1 * 16 + l16) * A_DIM + half * 16);
    v8f c0 = {}, c1 = {};
    c0 = __builtin_amdgcn_wmma_f32_16x16x32_bf16(false, aQ, false, *bp0,
                                                 (short)0, c0, false, false);
    c1 = __builtin_amdgcn_wmma_f32_16x16x32_bf16(false, aQ, false, *bp1,
                                                 (short)0, c1, false, false);
#pragma unroll
    for (int r = 0; r < 8; ++r) {
      sS[(half * 8 + r) * SS + kt * 16 + l16] = c0[r];
      rmax[r] = fmaxf(rmax[r], c0[r]);
    }
#pragma unroll
    for (int r = 0; r < 8; ++r) {
      sS[(half * 8 + r) * SS + kt1 * 16 + l16] = c1[r];
      rmax[r] = fmaxf(rmax[r], c1[r]);
    }
  }
  // Cross-lane max within each 16-lane half (the D-layout column group):
  // single ds_swizzle_b32 per exchange (group-of-32 XOR pattern, imm masks).
  SWIZ_MAX(0x041F)   // xor 1
  SWIZ_MAX(0x081F)   // xor 2
  SWIZ_MAX(0x101F)   // xor 4
  SWIZ_MAX(0x201F)   // xor 8
  if (l16 == 0) {                       // per-wave partial row maxima
#pragma unroll
    for (int r = 0; r < 8; ++r)
      sRed[(half * 8 + r) * 16 + wave] = rmax[r];
  }
  __syncthreads();
  if (tid < 16) {                       // final reduce over 8 waves
    float mx = sRed[tid * 16];
#pragma unroll
    for (int w = 1; w < 8; ++w) mx = fmaxf(mx, sRed[tid * 16 + w]);
    sMax[tid] = mx;
  }
  __syncthreads();

  // Phase 3: exp + row sum (16 rows x 4096), in LDS
  const int row = tid >> 4, t16 = tid & 15;
  {
    float rm = sMax[row], sum = 0.f;
    for (int j = 0; j < 256; ++j) {
      int idx = row * SS + t16 + j * 16;
      float e = __expf(sS[idx] - rm);
      sS[idx] = e;
      sum += e;
    }
    sRed[row * 16 + t16] = sum;
  }
  __syncthreads();
  if (t16 == 0) {
    float s = 0.f;
#pragma unroll
    for (int j = 0; j < 16; ++j) s += sRed[row * 16 + j];
    sInv[row] = 1.0f / s;
  }
  __syncthreads();

  // Phase 3.5: per row r (increasing): stage f32 row through registers,
  // stream normalized attention to HBM, then overwrite LDS bottom with bf16.
  // Alias safety: bf16 row r bytes [8208r, 8208r+8192) sit below f32 row r
  // start (16400r) for r>=1; row 0 handled by the read-all/barrier/write-all.
  {
    float* ab = attnOut + ((size_t)b * N_POS + q0) * N_POS;
    for (int r = 0; r < 16; ++r) {
      float iv = sInv[r];
      float st[16];
#pragma unroll
      for (int i = 0; i < 16; ++i) {
        float v = sS[r * SS + tid + i * 256] * iv;
        st[i] = v;
        ab[(size_t)r * N_POS + tid + i * 256] = v;   // coalesced HBM stream
      }
      __syncthreads();
#pragma unroll
      for (int i = 0; i < 16; ++i)
        sB[r * SSB + tid + i * 256] = (__bf16)st[i];
      __syncthreads();
    }
  }

  // Phase 4: out(16 x 256) = attn @ V ; each wave owns 2 column tiles of 16.
  // A-fragment comes straight from bf16 LDS: 2x ds_load_b128, no cvt.
  const float g = gammaPtr[0];
  for (int cc = 0; cc < 2; ++cc) {
    const int c0i = (wave * 2 + cc) * 16;
    const int ci  = c0i + l16;
    const __bf16* vb   = Vt + ((size_t)b * C_DIM + ci) * N_POS;
    const __bf16* arow = sB + l16 * SSB;
    v8f acc = {};
    for (int ks = 0; ks < 128; ++ks) {          // K-dim 4096 in steps of 32
      v16bf aS;
#pragma unroll
      for (int e = 0; e < 16; ++e) {
        int k = ks * 32 + half * 16 + e;
        aS[e] = arow[k];
      }
      const v16bf* bp = (const v16bf*)(vb + ks * 32 + half * 16);
      __builtin_prefetch(vb + (ks + 8) * 32, 0, 0);  // speculative V prefetch
      acc = __builtin_amdgcn_wmma_f32_16x16x32_bf16(false, aS, false, *bp,
                                                    (short)0, acc, false, false);
    }
    // Epilogue: out = gamma*acc + x, aligned float4 stores
    const int nq0 = q0 + half * 8;
    const float* xp = x   + ((size_t)b * C_DIM + ci) * N_POS + nq0;
    float*       op = out + ((size_t)b * C_DIM + ci) * N_POS + nq0;
    float4 x0 = *(const float4*)(xp);
    float4 x1 = *(const float4*)(xp + 4);
    float4 o0, o1;
    o0.x = g * acc[0] + x0.x; o0.y = g * acc[1] + x0.y;
    o0.z = g * acc[2] + x0.z; o0.w = g * acc[3] + x0.w;
    o1.x = g * acc[4] + x1.x; o1.y = g * acc[5] + x1.y;
    o1.z = g * acc[6] + x1.z; o1.w = g * acc[7] + x1.w;
    *(float4*)op       = o0;
    *(float4*)(op + 4) = o1;
  }
}

extern "C" void kernel_launch(void* const* d_in, const int* in_sizes, int n_in,
                              void* d_out, int out_size, void* d_ws, size_t ws_size,
                              hipStream_t stream) {
  const float* x  = (const float*)d_in[0];
  const float* Wq = (const float*)d_in[1];
  const float* bq = (const float*)d_in[2];
  const float* Wk = (const float*)d_in[3];
  const float* bk = (const float*)d_in[4];
  const float* Wv = (const float*)d_in[5];
  const float* bv = (const float*)d_in[6];
  const float* gm = (const float*)d_in[7];

  // workspace: bf16 Q (2MB) | bf16 K (2MB) | bf16 V^T (16MB)
  __bf16* Qb = (__bf16*)d_ws;
  __bf16* Kb = Qb + (size_t)8 * N_POS * A_DIM;
  __bf16* Vt = Kb + (size_t)8 * N_POS * A_DIM;

  float* out  = (float*)d_out;
  float* attn = out + OUT_ELEMS;

  proj32_kernel<<<dim3(128, 1), 256, 0, stream>>>(x, Wq, bq, Qb, 0);
  proj32_kernel<<<dim3(128, 1), 256, 0, stream>>>(x, Wk, bk, Kb, 0);
  proj32_kernel<<<dim3(128, 8), 256, 0, stream>>>(x, Wv, bv, Vt, 1);

  size_t smem = (size_t)16 * SS * 4 + 1024 + 256 * 4 + 16 * 4 + 16 * 4;
  (void)hipFuncSetAttribute(reinterpret_cast<const void*>(attn_kernel),
                            hipFuncAttributeMaxDynamicSharedMemorySize,
                            (int)smem);
  attn_kernel<<<dim3(2048), 256, smem, stream>>>(x, Qb, Kb, Vt, gm, out, attn);
}
